// DeformableCNNBlock_8272107012628
// MI455X (gfx1250) — matrix operand order repaired and checked by reading, not compiled, not run on hardware
//
#include <hip/hip_runtime.h>
#include <hip/hip_bf16.h>
#include <stdint.h>

// ---------------- problem constants ----------------
#define BATCH 8
#define CIN   64
#define COUT  128
#define HO    112
#define WO    112
#define HW    (HO * WO)                 // 12544 (multiple of 64)
#define NPOS  (BATCH * HW)              // 100352
#define KKTAP 9
#define KDIM  (CIN * KKTAP)             // 576
#define EPS   1e-5f

// LDS layout (dynamic shared): padded strides -> conflict-free ds_load_b128
#define BSTRIDE 584                     // 576 + 8 bf16 pad  (1168 B = 292 dw, 292%64=36)
#define ASTRIDE 72                      // 64 + 8 bf16 pad   (144 B  = 36 dw)
#define B_LDS_BYTES (COUT * BSTRIDE * 2)        // 149504
#define A_LDS_BYTES (64 * ASTRIDE * 2)          // 9216 per buffer
#define GEMM_LDS (B_LDS_BYTES + 2 * A_LDS_BYTES)// 167936

typedef __attribute__((ext_vector_type(16))) __bf16 v16bf;
typedef __attribute__((ext_vector_type(8)))  __bf16 v8bf;
typedef __attribute__((ext_vector_type(8)))  float  v8f;
typedef __attribute__((ext_vector_type(4)))  float  v4f;

// ---------------- kernel 1: NCHW -> NHWC transpose of x ----------------
__global__ __launch_bounds__(256) void k_transpose(const float* __restrict__ x,
                                                   float* __restrict__ xT) {
    size_t idx = (size_t)blockIdx.x * 256 + threadIdx.x;      // over B*C*H*W
    int w = (int)(idx % WO);
    int h = (int)((idx / WO) % HO);
    int c = (int)((idx / HW) % CIN);
    int b = (int)(idx / ((size_t)HW * CIN));
    xT[(((size_t)(b * HO + h) * WO) + w) * CIN + c] = x[idx];
}

// ---------------- kernel 2: repack w_dcn -> bf16 Bt[n][k], k = tap*64 + c ----------------
__global__ __launch_bounds__(256) void k_repack_w(const float* __restrict__ w,
                                                  __bf16* __restrict__ Bt) {
    int idx = blockIdx.x * 256 + threadIdx.x;                 // over 128*64*3*3 = 73728
    int kx = idx % 3;
    int ky = (idx / 3) % 3;
    int c  = (idx / 9) % CIN;
    int n  = idx / (9 * CIN);
    Bt[(size_t)n * KDIM + (ky * 3 + kx) * CIN + c] = (__bf16)w[idx];
}

// ---------------- kernel 3: offset conv (18 out ch, 3x3, pad 1) ----------------
__global__ __launch_bounds__(256) void k_off_conv(const float* __restrict__ xT,
                                                  const float* __restrict__ w_off,
                                                  const float* __restrict__ b_off,
                                                  float* __restrict__ off) {
    __shared__ float wl[KKTAP * CIN * 18];                    // [tap][c][o], 41472 B
    for (int idx = threadIdx.x; idx < KKTAP * CIN * 18; idx += 256) {
        // w_off layout: [o][c][ky][kx]
        int kx = idx % 3, ky = (idx / 3) % 3, c = (idx / 9) % CIN, o = idx / (9 * CIN);
        wl[((ky * 3 + kx) * CIN + c) * 18 + o] = w_off[idx];
    }
    __syncthreads();

    int g  = blockIdx.x * 256 + threadIdx.x;                  // over NPOS (exact)
    int wo = g % WO;
    int ho = (g / WO) % HO;
    int b  = g / HW;

    float acc[18];
#pragma unroll
    for (int o = 0; o < 18; ++o) acc[o] = b_off[o];

    for (int t = 0; t < KKTAP; ++t) {
        int yy = ho - 1 + t / 3;
        int xx = wo - 1 + t % 3;
        if (yy < 0 || yy >= HO || xx < 0 || xx >= WO) continue;
        const float* xp = xT + (((size_t)(b * HO + yy) * WO) + xx) * CIN;
        for (int c = 0; c < CIN; ++c) {
            float xv = xp[c];
            const float* wp = &wl[(t * CIN + c) * 18];
#pragma unroll
            for (int o = 0; o < 18; ++o) acc[o] += xv * wp[o];
        }
    }
#pragma unroll
    for (int o = 0; o < 18; ++o)
        off[((size_t)(b * 18 + o) * HW) + ho * WO + wo] = acc[o];
}

// ---------------- deformable bilinear sampling of one tap into padded LDS A ----------------
__device__ __forceinline__ void sample_tap(int t, int b, int ho, int wo, int i, int cg,
                                           const float* __restrict__ xT,
                                           const float* __restrict__ off,
                                           __bf16* __restrict__ Abuf) {
    const float dy = off[((size_t)(b * 18 + 2 * t + 0) * HW) + ho * WO + wo];
    const float dx = off[((size_t)(b * 18 + 2 * t + 1) * HW) + ho * WO + wo];
    const float py = (float)(ho - 1 + t / 3) + dy;
    const float px = (float)(wo - 1 + t % 3) + dx;
    const float fy = floorf(py), fx = floorf(px);
    const int   y0 = (int)fy, x0 = (int)fx;
    const float wy1 = py - fy, wx1 = px - fx;
    const float wy0 = 1.f - wy1, wx0 = 1.f - wx1;
    const int   c = cg * 16;                                  // 16 channels per thread

    v4f a0 = {}, a1 = {}, a2 = {}, a3 = {};
#pragma unroll
    for (int dyc = 0; dyc < 2; ++dyc) {
#pragma unroll
        for (int dxc = 0; dxc < 2; ++dxc) {
            const int  yi = y0 + dyc, xi = x0 + dxc;
            const bool valid = (yi >= 0) & (yi < HO) & (xi >= 0) & (xi < WO);
            const int  yc = min(max(yi, 0), HO - 1);
            const int  xc = min(max(xi, 0), WO - 1);
            const float wgt = (dyc ? wy1 : wy0) * (dxc ? wx1 : wx0) * (valid ? 1.f : 0.f);
            const float* p = xT + (((size_t)(b * HO + yc) * WO + xc) * CIN + c);
            v4f l0 = *(const v4f*)p;
            v4f l1 = *(const v4f*)(p + 4);
            v4f l2 = *(const v4f*)(p + 8);
            v4f l3 = *(const v4f*)(p + 12);
#pragma unroll
            for (int j = 0; j < 4; ++j) {
                a0[j] += wgt * l0[j];
                a1[j] += wgt * l1[j];
                a2[j] += wgt * l2[j];
                a3[j] += wgt * l3[j];
            }
        }
    }
    v8bf s0, s1;
#pragma unroll
    for (int j = 0; j < 4; ++j) {
        s0[j]     = (__bf16)a0[j];
        s0[j + 4] = (__bf16)a1[j];
        s1[j]     = (__bf16)a2[j];
        s1[j + 4] = (__bf16)a3[j];
    }
    *(v8bf*)&Abuf[i * ASTRIDE + c]     = s0;                  // ds_store_b128
    *(v8bf*)&Abuf[i * ASTRIDE + c + 8] = s1;
}

// ---------------- kernel 4: fused deformable sampling + WMMA GEMM ----------------
// GEMM: Y[m, n] = sum_k S[m, k] * Bt[n, k],  m = flat(b,ho,wo), n = cout, k = tap*64 + c
__global__ __launch_bounds__(256) void k_dcn_gemm(const float* __restrict__ xT,
                                                  const float* __restrict__ off,
                                                  const __bf16* __restrict__ Bt,
                                                  const float* __restrict__ bias,
                                                  float* __restrict__ y) {
    extern __shared__ __align__(16) char smem[];
    __bf16* Bl = (__bf16*)smem;                               // [128][BSTRIDE]
    __bf16* Al = (__bf16*)(smem + B_LDS_BYTES);               // 2 x [64][ASTRIDE]

    const int tid   = threadIdx.x;
    const int m0    = blockIdx.x * 64;
    const int wave  = tid >> 5;                               // wave32
    const int lane  = tid & 31;
    const int half  = lane >> 4;
    const int l16   = lane & 15;
    const int mwave = (wave >> 1) * 16;                       // 4 M-rows of waves
    const int nwave = (wave & 1) * 64;                        // 2 N-cols of waves

    // ---- stage full Bt into LDS with row padding, via async copy (ASYNCcnt) ----
    // 128 rows x 72 x 16B segments = 9216 transfers, 36 per thread
    {
#pragma unroll 4
        for (int it = 0; it < 36; ++it) {
            const int tr   = tid + it * 256;
            const int row  = tr / 72;
            const int seg  = tr % 72;
            const uint32_t goff = (uint32_t)(row * (KDIM * 2) + seg * 16);
            const uint32_t loff =
                (uint32_t)(uintptr_t)(&Bl[0]) + (uint32_t)(row * (BSTRIDE * 2) + seg * 16);
            asm volatile("global_load_async_to_lds_b128 %0, %1, %2 offset:0"
                         :
                         : "v"(loff), "v"(goff), "s"(Bt)
                         : "memory");
        }
    }

    v8f acc0 = {}, acc1 = {}, acc2 = {}, acc3 = {};

    // sampling-phase decomposition: 64 positions x 4 groups of 16 channels
    const int i  = tid & 63;
    const int cg = tid >> 6;
    const int g  = m0 + i;
    const int wo = g % WO;
    const int ho = (g / WO) % HO;
    const int b  = g / HW;

    // prologue: sample tap 0 into buffer 0 (overlaps with async B staging)
    sample_tap(0, b, ho, wo, i, cg, xT, off, Al);
    asm volatile("s_wait_asynccnt 0x0" ::: "memory");
    __syncthreads();

    // loop-invariant LDS addresses
    const __bf16* Arow0 = 0;
    const int a_off = (mwave + l16) * ASTRIDE + half * 8;     // + kc0 (+16 for hi chunk)
    (void)Arow0;

    for (int t = 0; t < KKTAP; ++t) {
        __bf16* Acur = Al + (t & 1) * (64 * ASTRIDE);
        __bf16* Anxt = Al + ((t + 1) & 1) * (64 * ASTRIDE);

        // sample tap t+1 (VMEM + VALU) — co-executes with the XDL WMMAs below
        if (t < KKTAP - 1)
            sample_tap(t + 1, b, ho, wo, i, cg, xT, off, Anxt);

        // ---- register-block the whole tap: preload A (2 frags) + B (8 frags),
        //      single dscnt wait, then 8 back-to-back WMMAs ----
        v16bf Afrag[2];
#pragma unroll
        for (int kc = 0; kc < 2; ++kc) {
            v8bf alo = *(const v8bf*)&Acur[a_off + kc * 32];
            v8bf ahi = *(const v8bf*)&Acur[a_off + kc * 32 + 16];
#pragma unroll
            for (int j = 0; j < 8; ++j) { Afrag[kc][j] = alo[j]; Afrag[kc][j + 8] = ahi[j]; }
        }

        v16bf Bfrag[4][2];
#pragma unroll
        for (int ng = 0; ng < 4; ++ng) {
            const int     n  = nwave + ng * 16 + l16;
            const __bf16* bp = &Bl[n * BSTRIDE + t * 64 + half * 8];
#pragma unroll
            for (int kc = 0; kc < 2; ++kc) {
                v8bf blo = *(const v8bf*)(bp + kc * 32);
                v8bf bhi = *(const v8bf*)(bp + kc * 32 + 16);
#pragma unroll
                for (int j = 0; j < 8; ++j) {
                    Bfrag[ng][kc][j]     = blo[j];
                    Bfrag[ng][kc][j + 8] = bhi[j];
                }
            }
        }

        // 8 WMMAs, interleaved so same-accumulator chains are 4 issues apart
#pragma unroll
        for (int kc = 0; kc < 2; ++kc) {
            acc0 = __builtin_amdgcn_wmma_f32_16x16x32_bf16(
                false, Afrag[kc], false, Bfrag[0][kc], (short)0, acc0, false, false);
            acc1 = __builtin_amdgcn_wmma_f32_16x16x32_bf16(
                false, Afrag[kc], false, Bfrag[1][kc], (short)0, acc1, false, false);
            acc2 = __builtin_amdgcn_wmma_f32_16x16x32_bf16(
                false, Afrag[kc], false, Bfrag[2][kc], (short)0, acc2, false, false);
            acc3 = __builtin_amdgcn_wmma_f32_16x16x32_bf16(
                false, Afrag[kc], false, Bfrag[3][kc], (short)0, acc3, false, false);
        }
        __syncthreads();
    }

    // ---- epilogue: tile is 64-aligned inside one image, so b/hw are block scalars;
    //      D rows M = v + 8*half are consecutive flat positions -> vector stores ----
    const int b_blk = m0 / HW;
    const int hw0   = (m0 % HW) + mwave + half * 8;
#pragma unroll
    for (int ng = 0; ng < 4; ++ng) {
        const v8f   acc = (ng == 0) ? acc0 : (ng == 1) ? acc1 : (ng == 2) ? acc2 : acc3;
        const int   n   = nwave + ng * 16 + l16;
        const float bn  = bias[n];
        float* yp = y + ((size_t)(b_blk * COUT + n)) * HW + hw0;
        v4f s0 = {acc[0] + bn, acc[1] + bn, acc[2] + bn, acc[3] + bn};
        v4f s1 = {acc[4] + bn, acc[5] + bn, acc[6] + bn, acc[7] + bn};
        *(v4f*)yp       = s0;                                 // global_store_b128
        *(v4f*)(yp + 4) = s1;
    }
}

// ---------------- kernel 5: per-channel batch stats (mean, biased var) ----------------
__global__ __launch_bounds__(256) void k_bn_stats(const float* __restrict__ y,
                                                  float* __restrict__ mean,
                                                  float* __restrict__ var) {
    __shared__ float ssum[256];
    __shared__ float ssq[256];
    const int n = blockIdx.x;                                 // channel
    float s = 0.f, q = 0.f;
    for (int g = threadIdx.x * 4; g < NPOS; g += 256 * 4) {
        int b  = g / HW;
        int hw = g % HW;                                      // HW % 4 == 0, no straddle
        v4f v  = *(const v4f*)&y[((size_t)(b * COUT + n)) * HW + hw];
#pragma unroll
        for (int j = 0; j < 4; ++j) { s += v[j]; q += v[j] * v[j]; }
    }
    ssum[threadIdx.x] = s;
    ssq[threadIdx.x]  = q;
    __syncthreads();
    for (int ofs = 128; ofs > 0; ofs >>= 1) {
        if (threadIdx.x < ofs) {
            ssum[threadIdx.x] += ssum[threadIdx.x + ofs];
            ssq[threadIdx.x]  += ssq[threadIdx.x + ofs];
        }
        __syncthreads();
    }
    if (threadIdx.x == 0) {
        float m = ssum[0] / (float)NPOS;
        mean[n] = m;
        var[n]  = ssq[0] / (float)NPOS - m * m;
    }
}

// ---------------- kernel 6: BN apply + ReLU, in place on d_out ----------------
__global__ __launch_bounds__(256) void k_bn_apply(float* __restrict__ y,
                                                  const float* __restrict__ mean,
                                                  const float* __restrict__ var,
                                                  const float* __restrict__ gamma,
                                                  const float* __restrict__ beta) {
    size_t idx = ((size_t)blockIdx.x * 256 + threadIdx.x) * 4; // HW % 4 == 0
    int    n   = (int)((idx / HW) % COUT);
    float  sc  = rsqrtf(var[n] + EPS) * gamma[n];
    float  sh  = beta[n] - mean[n] * sc;
    v4f v = *(const v4f*)&y[idx];
#pragma unroll
    for (int j = 0; j < 4; ++j) {
        float r = v[j] * sc + sh;
        v[j] = r > 0.f ? r : 0.f;
    }
    *(v4f*)&y[idx] = v;
}

// ---------------- launcher ----------------
extern "C" void kernel_launch(void* const* d_in, const int* in_sizes, int n_in,
                              void* d_out, int out_size, void* d_ws, size_t ws_size,
                              hipStream_t stream) {
    const float* x     = (const float*)d_in[0];   // [8,64,112,112]
    const float* w_off = (const float*)d_in[1];   // [18,64,3,3]
    const float* b_off = (const float*)d_in[2];   // [18]
    const float* w_dcn = (const float*)d_in[3];   // [128,64,3,3]
    const float* b_dcn = (const float*)d_in[4];   // [128]
    const float* gamma = (const float*)d_in[5];   // [128]
    const float* beta  = (const float*)d_in[6];   // [128]
    float*       yout  = (float*)d_out;           // [8,128,112,112]

    // workspace carve-up
    char*   ws   = (char*)d_ws;
    float*  xT   = (float*)ws;                                      // 25,690,112 B
    float*  off  = (float*)(ws + (size_t)NPOS * CIN * 4);           //  7,225,344 B
    __bf16* Bt   = (__bf16*)(ws + (size_t)NPOS * CIN * 4
                                + (size_t)NPOS * 18 * 4);           //    147,456 B
    float*  mean = (float*)((char*)Bt + (size_t)COUT * KDIM * 2);
    float*  var  = mean + COUT;

    // allow >64KB dynamic LDS for the GEMM kernel (deterministic host call)
    (void)hipFuncSetAttribute((const void*)k_dcn_gemm,
                              hipFuncAttributeMaxDynamicSharedMemorySize, GEMM_LDS);

    // 1) NCHW -> NHWC transpose of x  (6,422,528 = 25088*256)
    k_transpose<<<25088, 256, 0, stream>>>(x, xT);
    // 2) repack deform weights to bf16 [n][k]  (73728 = 288*256)
    k_repack_w<<<288, 256, 0, stream>>>(w_dcn, Bt);
    // 3) offset conv  (NPOS = 392*256)
    k_off_conv<<<392, 256, 0, stream>>>(xT, w_off, b_off, off);
    // 4) fused deformable-sample + WMMA GEMM  (NPOS/64 = 1568 tiles)
    k_dcn_gemm<<<1568, 256, GEMM_LDS, stream>>>(xT, off, Bt, b_dcn, yout);
    // 5) BN stats, one block per channel
    k_bn_stats<<<COUT, 256, 0, stream>>>(yout, mean, var);
    // 6) BN apply + ReLU in place  (12,845,056/4 = 12544*256)
    k_bn_apply<<<12544, 256, 0, stream>>>(yout, mean, var, gamma, beta);
}
